// AffinityPosWeight_4638564680209
// MI455X (gfx1250) — matrix operand order repaired and checked by reading, not compiled, not run on hardware
//
#include <hip/hip_runtime.h>

#define NTOK  8192
#define NMASK 8191
#define DDIM  32
#define PITCH 72        // ushorts per LDS x-tile row: 144B (16B-aligned, bank-padded)
#define PLSZ  (DDIM * PITCH)
#define KSLAB 64        // k-rows of x staged per buffer
#define BROWS 128       // output rows per workgroup

typedef __attribute__((ext_vector_type(16))) __bf16 v16bf;
typedef __attribute__((ext_vector_type(8)))  float  v8f;

union FragB16 { v16bf v; unsigned u[8]; };
union FragF32 { v8f   v; float    f[8]; };

// f32 -> bf16 bits, round-to-nearest-even
__device__ __forceinline__ unsigned f2bf(float f) {
  unsigned x = __float_as_uint(f);
  return (x + 0x7FFFu + ((x >> 16) & 1u)) >> 16;
}

// issue one 16B async global->LDS transfer (GVS mode: SGPR base + 32b lane offset)
#define ASYNC_LD_B128(LDSOFF, VOFF, SBASE)                                \
  asm volatile("global_load_async_to_lds_b128 %0, %1, %2"                 \
               :: "v"(LDSOFF), "v"(VOFF), "s"(SBASE) : "memory")

// ---------------- pre-pass: x (f32) -> transposed bf16 hi/lo planes in ws ----------------
// ws layout (ushort): batch b -> [hi: DDIM rows x NTOK][lo: DDIM rows x NTOK]
__global__ __launch_bounds__(256) void prep_planes(const float* __restrict__ x,
                                                   unsigned short* __restrict__ ws) {
  const int b  = blockIdx.y;
  const int t0 = (blockIdx.x * 256 + threadIdx.x) * 2;   // two tokens per thread
  const float* xb = x + (size_t)b * NTOK * DDIM;
  unsigned short* hp = ws + (size_t)b * 2 * NTOK * DDIM;
  unsigned short* lp = hp + (size_t)NTOK * DDIM;
  #pragma unroll 8
  for (int d = 0; d < DDIM; ++d) {
    float fa = xb[t0 * DDIM + d];
    float fb = xb[(t0 + 1) * DDIM + d];
    unsigned ha = f2bf(fa), hb = f2bf(fb);
    unsigned la = f2bf(fa - __uint_as_float(ha << 16));
    unsigned lb = f2bf(fb - __uint_as_float(hb << 16));
    *(unsigned*)(hp + (size_t)d * NTOK + t0) = ha | (hb << 16);
    *(unsigned*)(lp + (size_t)d * NTOK + t0) = la | (lb << 16);
  }
}

// ---------------- main: circulant x matmul via bf16x3 WMMA ----------------
__global__ __launch_bounds__(256) void circulant_bf16x3_wmma(
    const unsigned short* __restrict__ ws, const float* __restrict__ alpha,
    float* __restrict__ out) {
  // reversed alpha, packed (lo<<16)|hi bf16, 32-entry wrap margin: ~33 KB
  __shared__ __align__(16) unsigned       rev_pk[NTOK + 32];
  // double-buffered x slab tiles, FLAT array: [buf][hi/lo][row d][k], 18 KB.
  // (single shared array + integer offsets keeps every access in addrspace(3))
  __shared__ __align__(16) unsigned short xt[2 * 2 * PLSZ];

  const int tid   = threadIdx.x;
  const int lane  = tid & 31;
  const int wave  = tid >> 5;
  const int m     = lane & 15;   // row-in-tile (A) / column (B,C)
  const int hgrp  = lane >> 4;   // lane half -> K-offset group
  const int kb    = hgrp * 8;
  const int rw    = wave & 3;    // row-wave
  const int d0    = (wave >> 2) * 16;
  const int batch = blockIdx.y;
  const int rowBase = blockIdx.x * BROWS;

  const unsigned short* hp = ws + (size_t)batch * 2 * NTOK * DDIM;  // hi plane
  const unsigned short* lp = hp + (size_t)NTOK * DDIM;              // lo plane

  // ---- stage circulant generator: rev[t] = alpha[(-t) mod N], (lo<<16)|hi packed ----
  for (int t = tid; t < NTOK + 32; t += 256) {
    float f = alpha[(-t) & NMASK];
    unsigned hi = f2bf(f);
    unsigned lo = f2bf(f - __uint_as_float(hi << 16));
    rev_pk[t] = (lo << 16) | hi;
  }

  FragF32 acc[2][3];
  #pragma unroll
  for (int tt = 0; tt < 2; ++tt)
    #pragma unroll
    for (int p = 0; p < 3; ++p)
      #pragma unroll
      for (int r = 0; r < 8; ++r) acc[tt][p].f[r] = 0.f;

  // hoisted per-lane constants
  const int cA[2] = { -(rowBase + rw * 16 + m), -(rowBase + 64 + rw * 16 + m) };
  const int rr   = tid >> 3;                  // async: d-row 0..31
  const int part = tid & 7;                   // async: 16B chunk in row
  const unsigned gvoff = (unsigned)(rr * NTOK * 2 + part * 16);  // bytes
  const int ldsElem = rr * PITCH + part * 8;  // element offset within a plane
  unsigned lhs[2], lls[2];                    // LDS dest byte offsets per buffer
  lhs[0] = (unsigned)(uintptr_t)&xt[0 * 2 * PLSZ + ldsElem];
  lhs[1] = (unsigned)(uintptr_t)&xt[1 * 2 * PLSZ + ldsElem];
  lls[0] = (unsigned)(uintptr_t)&xt[0 * 2 * PLSZ + PLSZ + ldsElem];
  lls[1] = (unsigned)(uintptr_t)&xt[1 * 2 * PLSZ + PLSZ + ldsElem];
  const int bOff = (d0 + m) * PITCH + hgrp * 16;   // B-fragment element offset in plane

  // prologue: DMA slab 0 into buffer 0
  ASYNC_LD_B128(lhs[0], gvoff, (const void*)(hp + 0));
  ASYNC_LD_B128(lls[0], gvoff, (const void*)(lp + 0));

  for (int slab = 0; slab < NTOK; slab += KSLAB) {
    const int cur   = (slab >> 6) & 1;
    const int xbase = cur * 2 * PLSZ;           // buffer base (element offset)
    // DMA next slab into the other buffer, overlapping this slab's WMMAs
    if (slab + KSLAB < NTOK) {
      ASYNC_LD_B128(lhs[cur ^ 1], gvoff, (const void*)(hp + slab + KSLAB));
      ASYNC_LD_B128(lls[cur ^ 1], gvoff, (const void*)(lp + slab + KSLAB));
      asm volatile("s_wait_asynccnt 0x2" ::: "memory");  // current slab's 2 done
    } else {
      asm volatile("s_wait_asynccnt 0x0" ::: "memory");
    }
    __syncthreads();   // current buffer visible to all waves

    #pragma unroll
    for (int kk0 = 0; kk0 < KSLAB; kk0 += 32) {
      const int j0 = slab + kk0;

      // B fragments: lane = column, K striped 2-per-VGPR across lane halves.
      // Direct GEP on the shared array -> guaranteed ds_load_b128.
      FragB16 bhi, blo;
      {
        const uint4* ph = (const uint4*)&xt[xbase + bOff + kk0];
        const uint4* pl = (const uint4*)&xt[xbase + PLSZ + bOff + kk0];
        uint4 h0 = ph[0], h1 = ph[1], l0 = pl[0], l1 = pl[1];
        bhi.u[0] = h0.x; bhi.u[1] = h0.y; bhi.u[2] = h0.z; bhi.u[3] = h0.w;
        bhi.u[4] = h1.x; bhi.u[5] = h1.y; bhi.u[6] = h1.z; bhi.u[7] = h1.w;
        blo.u[0] = l0.x; blo.u[1] = l0.y; blo.u[2] = l0.z; blo.u[3] = l0.w;
        blo.u[4] = l1.x; blo.u[5] = l1.y; blo.u[6] = l1.z; blo.u[7] = l1.w;
      }

      #pragma unroll
      for (int tt = 0; tt < 2; ++tt) {
        // A fragments: circulant tile gathered from reversed alpha; packed dword
        // pairs split into (hi,hi+1)/(lo,lo+1) with single v_perm_b32 each
        FragB16 ahi, alo;
        const unsigned* rp = rev_pk + ((j0 + cA[tt]) & NMASK);
        #pragma unroll
        for (int v = 0; v < 8; ++v) {
          const int K = 2 * v + 2 * kb;
          unsigned p0 = rp[K], p1 = rp[K + 1];
          ahi.u[v] = __builtin_amdgcn_perm(p1, p0, 0x05040100u);
          alo.u[v] = __builtin_amdgcn_perm(p1, p0, 0x07060302u);
        }
        // bf16x3: hi*hi + hi*lo + lo*hi, independent f32 accumulator chains
        acc[tt][0].v = __builtin_amdgcn_wmma_f32_16x16x32_bf16(
            false, ahi.v, false, bhi.v, (short)0, acc[tt][0].v, false, false);
        acc[tt][1].v = __builtin_amdgcn_wmma_f32_16x16x32_bf16(
            false, ahi.v, false, blo.v, (short)0, acc[tt][1].v, false, false);
        acc[tt][2].v = __builtin_amdgcn_wmma_f32_16x16x32_bf16(
            false, alo.v, false, bhi.v, (short)0, acc[tt][2].v, false, false);
      }
    }
    __syncthreads();   // all waves done reading buf before it is DMA-overwritten
  }

  // C/D layout: VGPR r -> row M = r + 8*hgrp, column n = m
  #pragma unroll
  for (int tt = 0; tt < 2; ++tt) {
    float* ob = out + ((size_t)batch * NTOK + rowBase + tt * 64 + rw * 16 + 8 * hgrp) * DDIM
                    + d0 + m;
    #pragma unroll
    for (int r = 0; r < 8; ++r)
      ob[r * DDIM] = acc[tt][0].f[r] + acc[tt][1].f[r] + acc[tt][2].f[r];
  }
}

extern "C" void kernel_launch(void* const* d_in, const int* in_sizes, int n_in,
                              void* d_out, int out_size, void* d_ws, size_t ws_size,
                              hipStream_t stream) {
  const float* x     = (const float*)d_in[0];
  const float* alpha = (const float*)d_in[1];
  float* out         = (float*)d_out;
  unsigned short* ws = (unsigned short*)d_ws;      // needs 2 MB (B*2*NTOK*DDIM ushorts)
  int nbatch = in_sizes[0] / (NTOK * DDIM);        // B = 2

  dim3 gprep(NTOK / 512, nbatch);                  // (16, 2)
  prep_planes<<<gprep, 256, 0, stream>>>(x, ws);

  dim3 grid(NTOK / BROWS, nbatch);                 // (64, 2)
  circulant_bf16x3_wmma<<<grid, 256, 0, stream>>>(ws, alpha, out);
}
